// MultiModelFusionSystem_61469571940557
// MI455X (gfx1250) — matrix operand order repaired and checked by reading, not compiled, not run on hardware
//
#include <hip/hip_runtime.h>
#include <hip/hip_bf16.h>

// ---------------------------------------------------------------------------
// MultiModelFusionSystem on MI455X (gfx1250), wave32 + WMMA bf16.
//
// Precision: bf16 inputs to v_wmma_f32_16x16x32_bf16, f32 accumulation.
// Softmax: reference clips logits to +-10 (cross & memory attention), so
// exp() is bounded and we skip max-subtraction entirely (one-pass denom).
// Self-attention logits are tiny (0.02-scale weights) -> same trick.
// LDS tile staging uses GLOBAL_LOAD_ASYNC_TO_LDS_B128 (ASYNCcnt path) so the
// copy bypasses VGPRs; it is waited on before wave0's score WMMAs start.
// GEMM keeps a branch-free inner loop: clause-grouped global_load_b128 feeding
// back-to-back WMMAs under partial s_wait_loadcnt (operands are L2-resident).
// ---------------------------------------------------------------------------

typedef __bf16 bf16_t;
typedef bf16_t v8bf  __attribute__((ext_vector_type(8)));
typedef bf16_t v16bf __attribute__((ext_vector_type(16)));
typedef float  v8f   __attribute__((ext_vector_type(8)));

#define E_ 4
#define N_ 2048
#define D_ 512
#define M_ 2048
#define L_ 3
#define F_ 2048
#define H_ 8
#define EN_ (E_ * N_)
#define INV_TEMP 0.044194173824159216f   /* 1/sqrt(512) */
#define INV_TEMP_H 0.125f                /* 1/sqrt(64)  */

// ------------------------- WMMA fragment helpers ---------------------------
// A fragment: 16x32 (MxK) bf16.  lane&15 = row; lane>>4 = K half:
//   elements 0..15 -> K = (lane>>4)*16 + e
__device__ __forceinline__ v16bf load_a_frag(const bf16_t* base, int ld, int k0, int lane) {
  const bf16_t* p = base + (long)(lane & 15) * ld + k0 + ((lane >> 4) << 4);
  v8bf lo = *(const v8bf*)p;
  v8bf hi = *(const v8bf*)(p + 8);
  v16bf r;
#pragma unroll
  for (int e = 0; e < 8; ++e) { r[e] = lo[e]; r[e + 8] = hi[e]; }
  return r;
}

// B fragment: 32x16 (KxN) bf16, source stored row-major as [N, K] (i.e. B^T):
//   lane&15 = column n; lane>>4 = K half (0..15 / 16..31); elements K-contig.
__device__ __forceinline__ v16bf load_b_frag(const bf16_t* base, int ld, int k0, int lane) {
  const bf16_t* p = base + (long)(lane & 15) * ld + k0 + ((lane >> 4) << 4);
  v8bf lo = *(const v8bf*)p;
  v8bf hi = *(const v8bf*)(p + 8);
  v16bf r;
#pragma unroll
  for (int e = 0; e < 8; ++e) { r[e] = lo[e]; r[e + 8] = hi[e]; }
  return r;
}

// B fragment from a row-major [K, N] tile (e.g. V tile in LDS): strided gather.
__device__ __forceinline__ v16bf load_b_strided(const bf16_t* base, int ld, int lane) {
  const bf16_t* p = base + (lane & 15) + (long)((lane >> 4) << 4) * ld;
  v16bf r;
#pragma unroll
  for (int e = 0; e < 16; ++e) r[e] = p[(long)e * ld];
  return r;
}

__device__ __forceinline__ v8f wmma_bf16(v16bf a, v16bf b, v8f c) {
  return __builtin_amdgcn_wmma_f32_16x16x32_bf16(false, a, false, b, (short)0, c, false, false);
}

__device__ __forceinline__ float half_reduce16(float v) {
  v += __shfl_xor(v, 1, 32);
  v += __shfl_xor(v, 2, 32);
  v += __shfl_xor(v, 4, 32);
  v += __shfl_xor(v, 8, 32);
  return v;
}

__device__ __forceinline__ float clip10(float x) { return fminf(fmaxf(x, -10.f), 10.f); }

// ---------------- async global->LDS copy (CDNA5 ASYNCcnt path) -------------
// 16 bytes per lane, no VGPR round-trip; flat LDS aperture truncates to the
// 32-bit LDS offset (ISA 10.2).
__device__ __forceinline__ void async_copy16(const bf16_t* __restrict__ g, bf16_t* l) {
  unsigned int loff = (unsigned int)(unsigned long long)l;
  unsigned long long ga = (unsigned long long)g;
  asm volatile("global_load_async_to_lds_b128 %0, %1, off"
               :: "v"(loff), "v"(ga) : "memory");
}
__device__ __forceinline__ void async_wait0() {
  asm volatile("s_wait_asynccnt 0" ::: "memory");
}

// ------------------------------ elementwise --------------------------------
__global__ void k_cvt_bf16(const float* __restrict__ in, bf16_t* __restrict__ out, long n) {
  long i = blockIdx.x * 256L + threadIdx.x;
  long stride = (long)gridDim.x * 256L;
  for (; i < n; i += stride) out[i] = (bf16_t)in[i];
}

// LayerNorm over rows of width D_; optional residual, optional affine
// (g/b selected per group of rows: g_eff = g + (row/rows_per_group)*gstride).
__global__ void k_layernorm(const float* __restrict__ x, const float* __restrict__ res,
                            const float* __restrict__ g, const float* __restrict__ b,
                            int rows_per_group, long gstride,
                            float* __restrict__ y, bf16_t* __restrict__ ybf) {
  __shared__ float red[256];
  const int row = blockIdx.x;
  const float* xr = x + (long)row * D_;
  const float* rr = res ? res + (long)row * D_ : nullptr;
  float s = 0.f;
  for (int d = threadIdx.x; d < D_; d += 256) {
    float v = xr[d] + (rr ? rr[d] : 0.f);
    s += v;
  }
  red[threadIdx.x] = s; __syncthreads();
  for (int k = 128; k > 0; k >>= 1) {
    if (threadIdx.x < k) red[threadIdx.x] += red[threadIdx.x + k];
    __syncthreads();
  }
  float mean = red[0] * (1.f / D_);
  __syncthreads();
  float s2 = 0.f;
  for (int d = threadIdx.x; d < D_; d += 256) {
    float v = xr[d] + (rr ? rr[d] : 0.f) - mean;
    s2 += v * v;
  }
  red[threadIdx.x] = s2; __syncthreads();
  for (int k = 128; k > 0; k >>= 1) {
    if (threadIdx.x < k) red[threadIdx.x] += red[threadIdx.x + k];
    __syncthreads();
  }
  float rstd = rsqrtf(red[0] * (1.f / D_) + 1e-5f);
  const float* ge = g ? g + (long)(row / rows_per_group) * gstride : nullptr;
  const float* be = b ? b + (long)(row / rows_per_group) * gstride : nullptr;
  for (int d = threadIdx.x; d < D_; d += 256) {
    float v = (xr[d] + (rr ? rr[d] : 0.f) - mean) * rstd;
    if (ge) v = v * ge[d] + be[d];
    long idx = (long)row * D_ + d;
    if (y)   y[idx] = v;
    if (ybf) ybf[idx] = (bf16_t)v;
  }
}

// ------------------------------- WMMA GEMM ---------------------------------
// C[M,N] = act(A[M,K] @ B[N,K]^T + bias[n]) (+ res).  All bf16 in, f32 acc.
// grid: (M/32, N/64, batch); block 256 = 8 waves, each wave a 16x16 tile.
// Operands are L2-resident (192 MB) -> direct global loads, branch-free loop
// so the compiler pipelines load clauses against WMMAs (s_wait_loadcnt > 0).
__global__ void k_gemm(const bf16_t* __restrict__ A, int lda, long strideA,
                       const bf16_t* __restrict__ B, int ldb, long strideB,
                       const float* __restrict__ bias, long strideBias,
                       const float* __restrict__ res, long strideRes,
                       float* __restrict__ Cf, bf16_t* __restrict__ Cb,
                       int ldc, long strideC, int K, int relu) {
  const int lane = threadIdx.x & 31, w = threadIdx.x >> 5;
  const int z = blockIdx.z;
  const int m0 = blockIdx.x * 32 + (w & 1) * 16;
  const int n0 = blockIdx.y * 64 + (w >> 1) * 16;
  const bf16_t* Az = A + (long)z * strideA + (long)m0 * lda;
  const bf16_t* Bz = B + (long)z * strideB + (long)n0 * ldb;
  v8f acc = {};
  for (int k0 = 0; k0 < K; k0 += 32) {
    v16bf a = load_a_frag(Az, lda, k0, lane);
    v16bf b = load_b_frag(Bz, ldb, k0, lane);
    acc = wmma_bf16(a, b, acc);
  }
  const int col = n0 + (lane & 15);
  const int rbase = m0 + (lane >> 4) * 8;
  float bb = bias ? bias[(long)z * strideBias + col] : 0.f;
#pragma unroll
  for (int r = 0; r < 8; ++r) {
    float v = acc[r] + bb;
    if (relu) v = fmaxf(v, 0.f);
    long idx = (long)(rbase + r) * ldc + col;
    if (res) v += res[(long)z * strideRes + idx];
    long cidx = (long)z * strideC + idx;
    if (Cf) Cf[cidx] = v;
    if (Cb) Cb[cidx] = (bf16_t)v;
  }
}

// --------------------- cross-attention (8192 keys, D=512) ------------------
// Pass A: S[q] = sum_m exp(clip(q.k/temp)), skipping own expert block.
__global__ void k_cross_denom(const bf16_t* __restrict__ X, float* __restrict__ S) {
  const int lane = threadIdx.x & 31, w = threadIdx.x >> 5;
  const int q0 = (blockIdx.x * 8 + w) * 16;
  if (q0 >= EN_) return;
  const int e = q0 >> 11;
  const int own0 = e << 11, own1 = own0 + N_;
  v16bf afr[16];
#pragma unroll
  for (int kk = 0; kk < 16; ++kk)
    afr[kk] = load_a_frag(X + (long)q0 * D_, D_, kk * 32, lane);
  float ar[8] = {};
  for (int m0 = 0; m0 < EN_; m0 += 16) {
    if (m0 >= own0 && m0 < own1) continue;
    v8f c = {};
#pragma unroll
    for (int kk = 0; kk < 16; ++kk)
      c = wmma_bf16(afr[kk], load_b_frag(X + (long)m0 * D_, D_, kk * 32, lane), c);
#pragma unroll
    for (int r = 0; r < 8; ++r) ar[r] += __expf(clip10(c[r] * INV_TEMP));
  }
#pragma unroll
  for (int r = 0; r < 8; ++r) {
    float v = half_reduce16(ar[r]);
    if ((lane & 15) == 0) S[q0 + (lane >> 4) * 8 + r] = v;
  }
}

// Pass B: expert_in = ln_tok + (sum_m p * flat[m]) / S.  Block: 16q x 512d.
__global__ void k_cross_apply(const bf16_t* __restrict__ X, const float* __restrict__ S,
                              const float* __restrict__ lnf,
                              float* __restrict__ eif, bf16_t* __restrict__ eib) {
  __shared__ bf16_t pt[16 * 32];
  __shared__ bf16_t vt[32 * D_];
  const int lane = threadIdx.x & 31, w = threadIdx.x >> 5;
  const int q0 = blockIdx.x * 16;
  const int e = q0 >> 11;
  const int own0 = e << 11, own1 = own0 + N_;
  const int dcol0 = w * 64;
  v16bf afr[16];
  if (w == 0) {
#pragma unroll
    for (int kk = 0; kk < 16; ++kk)
      afr[kk] = load_a_frag(X + (long)q0 * D_, D_, kk * 32, lane);
  }
  v8f acc[4] = {};
  for (int m0 = 0; m0 < EN_; m0 += 32) {
    if (m0 >= own0 && m0 < own1) continue;
    const bf16_t* src = X + (long)m0 * D_;
    for (int t = threadIdx.x; t < 32 * D_ / 8; t += 256)
      async_copy16(src + (long)t * 8, vt + (long)t * 8);
    async_wait0();
    __syncthreads();
    if (w == 0) {
#pragma unroll
      for (int half = 0; half < 2; ++half) {
        v8f c = {};
#pragma unroll
        for (int kk = 0; kk < 16; ++kk)
          c = wmma_bf16(afr[kk], load_b_frag(vt + (long)half * 16 * D_, D_, kk * 32, lane), c);
        int col = lane & 15, hi = lane >> 4;
#pragma unroll
        for (int r = 0; r < 8; ++r)
          pt[(hi * 8 + r) * 32 + half * 16 + col] = (bf16_t)__expf(clip10(c[r] * INV_TEMP));
      }
    }
    __syncthreads();
    v16bf a = load_a_frag(pt, 32, 0, lane);
#pragma unroll
    for (int j = 0; j < 4; ++j)
      acc[j] = wmma_bf16(a, load_b_strided(vt + dcol0 + j * 16, D_, lane), acc[j]);
    __syncthreads();
  }
  const int col = lane & 15, rbase = (lane >> 4) * 8;
#pragma unroll
  for (int j = 0; j < 4; ++j)
#pragma unroll
    for (int r = 0; r < 8; ++r) {
      int q = q0 + rbase + r;
      int d = dcol0 + j * 16 + col;
      long idx = (long)q * D_ + d;
      float v = acc[j][r] / S[q] + lnf[idx];
      eif[idx] = v;
      eib[idx] = (bf16_t)v;
    }
}

// --------------------- encoder self-attention (dh=64) ----------------------
__global__ void k_self_denom(const bf16_t* __restrict__ qb, const bf16_t* __restrict__ kb,
                             float* __restrict__ S) {
  const int lane = threadIdx.x & 31, w = threadIdx.x >> 5;
  const int eh = blockIdx.y, e = eh >> 3, h = eh & 7;
  const int q0 = (blockIdx.x * 8 + w) * 16;
  const long base = (long)e * N_ * D_ + h * 64;
  v16bf a0 = load_a_frag(qb + base + (long)q0 * D_, D_, 0, lane);
  v16bf a1 = load_a_frag(qb + base + (long)q0 * D_, D_, 32, lane);
  float ar[8] = {};
  for (int m0 = 0; m0 < N_; m0 += 16) {
    v8f c = {};
    c = wmma_bf16(a0, load_b_frag(kb + base + (long)m0 * D_, D_, 0, lane), c);
    c = wmma_bf16(a1, load_b_frag(kb + base + (long)m0 * D_, D_, 32, lane), c);
#pragma unroll
    for (int r = 0; r < 8; ++r) ar[r] += __expf(c[r] * INV_TEMP_H);
  }
#pragma unroll
  for (int r = 0; r < 8; ++r) {
    float v = half_reduce16(ar[r]);
    if ((lane & 15) == 0) S[(long)eh * N_ + q0 + (lane >> 4) * 8 + r] = v;
  }
}

__global__ void k_self_apply(const bf16_t* __restrict__ qb, const bf16_t* __restrict__ kb,
                             const bf16_t* __restrict__ vb, const float* __restrict__ S,
                             bf16_t* __restrict__ ob) {
  __shared__ bf16_t pt[16 * 32];
  __shared__ bf16_t vt[32 * 64];
  const int lane = threadIdx.x & 31, w = threadIdx.x >> 5;  // 4 waves
  const int eh = blockIdx.y, e = eh >> 3, h = eh & 7;
  const int q0 = blockIdx.x * 16;
  const long base = (long)e * N_ * D_ + h * 64;
  const int dcol = w * 16;
  v16bf a0, a1;
  if (w == 0) {
    a0 = load_a_frag(qb + base + (long)q0 * D_, D_, 0, lane);
    a1 = load_a_frag(qb + base + (long)q0 * D_, D_, 32, lane);
  }
  v8f acc = {};
  for (int m0 = 0; m0 < N_; m0 += 32) {
    for (int t = threadIdx.x; t < 256; t += 128) {
      int row = t >> 3, c8 = (t & 7) << 3;
      async_copy16(vb + base + (long)(m0 + row) * D_ + c8, vt + row * 64 + c8);
    }
    async_wait0();
    __syncthreads();
    if (w == 0) {
#pragma unroll
      for (int half = 0; half < 2; ++half) {
        v8f c = {};
        c = wmma_bf16(a0, load_b_frag(kb + base + (long)(m0 + half * 16) * D_, D_, 0, lane), c);
        c = wmma_bf16(a1, load_b_frag(kb + base + (long)(m0 + half * 16) * D_, D_, 32, lane), c);
        int col = lane & 15, hi = lane >> 4;
#pragma unroll
        for (int r = 0; r < 8; ++r)
          pt[(hi * 8 + r) * 32 + half * 16 + col] = (bf16_t)__expf(c[r] * INV_TEMP_H);
      }
    }
    __syncthreads();
    v16bf a = load_a_frag(pt, 32, 0, lane);
    acc = wmma_bf16(a, load_b_strided(vt + dcol, 64, lane), acc);
    __syncthreads();
  }
  const int col = lane & 15, rbase = (lane >> 4) * 8;
#pragma unroll
  for (int r = 0; r < 8; ++r) {
    int q = q0 + rbase + r;
    float v = acc[r] / S[(long)eh * N_ + q];
    ob[base + (long)q * D_ + dcol + col] = (bf16_t)v;
  }
}

// ----------------------- memory attention (M=2048) -------------------------
__global__ void k_mem_denom(const bf16_t* __restrict__ eo, const bf16_t* __restrict__ mem,
                            float* __restrict__ S) {
  const int lane = threadIdx.x & 31, w = threadIdx.x >> 5;
  const int el = blockIdx.y, e = el / L_, l = el % L_;
  const int q0 = (blockIdx.x * 8 + w) * 16;
  v16bf afr[16];
#pragma unroll
  for (int kk = 0; kk < 16; ++kk)
    afr[kk] = load_a_frag(eo + ((long)e * N_ + q0) * D_, D_, kk * 32, lane);
  float ar[8] = {};
  for (int m0 = 0; m0 < M_; m0 += 16) {
    v8f c = {};
#pragma unroll
    for (int kk = 0; kk < 16; ++kk)
      c = wmma_bf16(afr[kk], load_b_frag(mem + ((long)l * M_ + m0) * D_, D_, kk * 32, lane), c);
#pragma unroll
    for (int r = 0; r < 8; ++r) ar[r] += __expf(clip10(c[r] * INV_TEMP));
  }
#pragma unroll
  for (int r = 0; r < 8; ++r) {
    float v = half_reduce16(ar[r]);
    if ((lane & 15) == 0) S[(long)el * N_ + q0 + (lane >> 4) * 8 + r] = v;
  }
}

// retrieval -> cat[e, n, l*D + d] (bf16) for the agg GEMM
__global__ void k_mem_apply(const bf16_t* __restrict__ eo, const bf16_t* __restrict__ mem,
                            const float* __restrict__ S, bf16_t* __restrict__ cat) {
  __shared__ bf16_t pt[16 * 32];
  __shared__ bf16_t vt[32 * D_];
  const int lane = threadIdx.x & 31, w = threadIdx.x >> 5;
  const int el = blockIdx.y, e = el / L_, l = el % L_;
  const int q0 = blockIdx.x * 16;
  const int dcol0 = w * 64;
  v16bf afr[16];
  if (w == 0) {
#pragma unroll
    for (int kk = 0; kk < 16; ++kk)
      afr[kk] = load_a_frag(eo + ((long)e * N_ + q0) * D_, D_, kk * 32, lane);
  }
  v8f acc[4] = {};
  for (int m0 = 0; m0 < M_; m0 += 32) {
    const bf16_t* src = mem + ((long)l * M_ + m0) * D_;
    for (int t = threadIdx.x; t < 32 * D_ / 8; t += 256)
      async_copy16(src + (long)t * 8, vt + (long)t * 8);
    async_wait0();
    __syncthreads();
    if (w == 0) {
#pragma unroll
      for (int half = 0; half < 2; ++half) {
        v8f c = {};
#pragma unroll
        for (int kk = 0; kk < 16; ++kk)
          c = wmma_bf16(afr[kk], load_b_frag(vt + (long)half * 16 * D_, D_, kk * 32, lane), c);
        int col = lane & 15, hi = lane >> 4;
#pragma unroll
        for (int r = 0; r < 8; ++r)
          pt[(hi * 8 + r) * 32 + half * 16 + col] = (bf16_t)__expf(clip10(c[r] * INV_TEMP));
      }
    }
    __syncthreads();
    v16bf a = load_a_frag(pt, 32, 0, lane);
#pragma unroll
    for (int j = 0; j < 4; ++j)
      acc[j] = wmma_bf16(a, load_b_strided(vt + dcol0 + j * 16, D_, lane), acc[j]);
    __syncthreads();
  }
  const int col = lane & 15, rbase = (lane >> 4) * 8;
#pragma unroll
  for (int j = 0; j < 4; ++j)
#pragma unroll
    for (int r = 0; r < 8; ++r) {
      int q = q0 + rbase + r;
      int d = dcol0 + j * 16 + col;
      float v = acc[j][r] / S[(long)el * N_ + q];
      cat[((long)e * N_ + q) * (L_ * D_) + l * D_ + d] = (bf16_t)v;
    }
}

// memory update: new = 0.9*mem + clip(0.1 * sum_e clip(attn_e^T @ eo_e, +-1), +-0.1)
__global__ void k_mem_update(const bf16_t* __restrict__ mem_bf, const float* __restrict__ mem_f,
                             const bf16_t* __restrict__ eo, const float* __restrict__ S,
                             float* __restrict__ newmem) {
  __shared__ bf16_t pt[16 * 32];
  __shared__ bf16_t vt[32 * D_];
  const int lane = threadIdx.x & 31, w = threadIdx.x >> 5;
  const int m0 = blockIdx.x * 16;
  const int l = blockIdx.y;
  const int dcol0 = w * 64;
  v16bf afr[16];
  if (w == 0) {
#pragma unroll
    for (int kk = 0; kk < 16; ++kk)
      afr[kk] = load_a_frag(mem_bf + ((long)l * M_ + m0) * D_, D_, kk * 32, lane);
  }
  v8f tot[4] = {};
  for (int e = 0; e < E_; ++e) {
    v8f acc[4] = {};
    for (int n0 = 0; n0 < N_; n0 += 32) {
      const bf16_t* src = eo + ((long)e * N_ + n0) * D_;
      for (int t = threadIdx.x; t < 32 * D_ / 8; t += 256)
        async_copy16(src + (long)t * 8, vt + (long)t * 8);
      async_wait0();
      __syncthreads();
      if (w == 0) {
#pragma unroll
        for (int half = 0; half < 2; ++half) {
          v8f c = {};
#pragma unroll
          for (int kk = 0; kk < 16; ++kk)
            c = wmma_bf16(afr[kk], load_b_frag(vt + (long)half * 16 * D_, D_, kk * 32, lane), c);
          int col = lane & 15, hi = lane >> 4;
          float sc = 1.f / S[((long)e * L_ + l) * N_ + n0 + half * 16 + col];
#pragma unroll
          for (int r = 0; r < 8; ++r)
            pt[(hi * 8 + r) * 32 + half * 16 + col] =
                (bf16_t)(__expf(clip10(c[r] * INV_TEMP)) * sc);
        }
      }
      __syncthreads();
      v16bf a = load_a_frag(pt, 32, 0, lane);
#pragma unroll
      for (int j = 0; j < 4; ++j)
        acc[j] = wmma_bf16(a, load_b_strided(vt + dcol0 + j * 16, D_, lane), acc[j]);
      __syncthreads();
    }
#pragma unroll
    for (int j = 0; j < 4; ++j)
#pragma unroll
      for (int r = 0; r < 8; ++r) tot[j][r] += fminf(fmaxf(acc[j][r], -1.f), 1.f);
  }
  const int col = lane & 15, rbase = (lane >> 4) * 8;
#pragma unroll
  for (int j = 0; j < 4; ++j)
#pragma unroll
    for (int r = 0; r < 8; ++r) {
      int m = m0 + rbase + r;
      int d = dcol0 + j * 16 + col;
      long idx = ((long)l * M_ + m) * D_ + d;
      float u = fminf(fmaxf(0.1f * tot[j][r], -0.1f), 0.1f);
      newmem[idx] = 0.9f * mem_f[idx] + u;
    }
}

// ------------------------------ epilogue -----------------------------------
__global__ void k_pooled(const float* __restrict__ out_f, float* __restrict__ pooled) {
  int idx = blockIdx.x * 256 + threadIdx.x;
  if (idx >= E_ * D_) return;
  int e = idx >> 9, d = idx & (D_ - 1);
  const float* p = out_f + (long)e * N_ * D_ + d;
  float s = 0.f;
  for (int n = 0; n < N_; ++n) s += p[(long)n * D_];
  pooled[idx] = s * (1.f / N_);
}

__global__ void k_gate(const float* __restrict__ pooled, const float* __restrict__ gate_w,
                       float* __restrict__ gate_out) {
  __shared__ float red[256];
  __shared__ float logits[E_];
  for (int e = 0; e < E_; ++e) {
    float s = 0.f;
    for (int d = threadIdx.x; d < D_; d += 256) s += pooled[e * D_ + d] * gate_w[d];
    red[threadIdx.x] = s; __syncthreads();
    for (int k = 128; k > 0; k >>= 1) {
      if (threadIdx.x < k) red[threadIdx.x] += red[threadIdx.x + k];
      __syncthreads();
    }
    if (threadIdx.x == 0) logits[e] = clip10(red[0]);
    __syncthreads();
  }
  if (threadIdx.x == 0) {
    float mx = -1e30f;
    for (int e = 0; e < E_; ++e) mx = fmaxf(mx, logits[e]);
    float den = 0.f;
    for (int e = 0; e < E_; ++e) den += __expf(logits[e] - mx);
    for (int e = 0; e < E_; ++e) gate_out[e] = __expf(logits[e] - mx) / den;
  }
}

__global__ void k_fused(const float* __restrict__ out_f, const float* __restrict__ gate,
                        float* __restrict__ fused) {
  const long total = (long)N_ * D_;
  long i = blockIdx.x * 256L + threadIdx.x;
  long stride = (long)gridDim.x * 256L;
  for (; i < total; i += stride) {
    float s = 0.f;
#pragma unroll
    for (int e = 0; e < E_; ++e) s += gate[e] * out_f[(long)e * total + i];
    fused[i] = s;
  }
}

// ------------------------------- launcher ----------------------------------
extern "C" void kernel_launch(void* const* d_in, const int* in_sizes, int n_in,
                              void* d_out, int out_size, void* d_ws, size_t ws_size,
                              hipStream_t stream) {
  (void)in_sizes; (void)n_in; (void)out_size; (void)ws_size;
  const float* tokens   = (const float*)d_in[0];
  const float* wqkv     = (const float*)d_in[1];
  const float* bqkv     = (const float*)d_in[2];
  const float* wo       = (const float*)d_in[3];
  const float* bo       = (const float*)d_in[4];
  const float* ln_g     = (const float*)d_in[5];
  const float* ln_b     = (const float*)d_in[6];
  const float* w1       = (const float*)d_in[7];
  const float* b1       = (const float*)d_in[8];
  const float* w2       = (const float*)d_in[9];
  const float* b2       = (const float*)d_in[10];
  const float* memories = (const float*)d_in[11];
  const float* agg_w    = (const float*)d_in[12];
  const float* agg_b    = (const float*)d_in[13];
  const float* gate_w   = (const float*)d_in[14];

  float* out_fused  = (float*)d_out;
  float* out_gate   = out_fused + (long)N_ * D_;
  float* out_newmem = out_gate + E_;

  // ---- workspace bump allocator (lifetime-based aliasing) ----
  char* wsp = (char*)d_ws;
  auto alloc = [&](size_t bytes) -> void* {
    void* p = wsp;
    wsp += (bytes + 255) & ~(size_t)255;
    return p;
  };
  float*  ln_f   = (float*) alloc((size_t)EN_ * D_ * 4);
  bf16_t* ln_bf  = (bf16_t*)alloc((size_t)EN_ * D_ * 2);
  float*  Scr    = (float*) alloc((size_t)EN_ * 4);
  float*  ei_f   = (float*) alloc((size_t)EN_ * D_ * 4);
  bf16_t* ei_bf  = (bf16_t*)alloc((size_t)EN_ * D_ * 2);
  bf16_t* wqkv_b = (bf16_t*)alloc((size_t)E_ * 3 * D_ * D_ * 2);
  bf16_t* wo_b   = (bf16_t*)alloc((size_t)E_ * D_ * D_ * 2);
  bf16_t* w1_b   = (bf16_t*)alloc((size_t)E_ * F_ * D_ * 2);
  bf16_t* w2_b   = (bf16_t*)alloc((size_t)E_ * D_ * F_ * 2);
  bf16_t* aggw_b = (bf16_t*)alloc((size_t)D_ * L_ * D_ * 2);
  bf16_t* mem_b  = (bf16_t*)alloc((size_t)L_ * M_ * D_ * 2);
  bf16_t* q_b    = (bf16_t*)alloc((size_t)EN_ * D_ * 2);
  bf16_t* k_b    = (bf16_t*)alloc((size_t)EN_ * D_ * 2);
  bf16_t* v_b    = (bf16_t*)alloc((size_t)EN_ * D_ * 2);
  float*  Sat    = (float*) alloc((size_t)E_ * H_ * N_ * 4);
  float*  x1_f   = (float*) alloc((size_t)EN_ * D_ * 4);
  bf16_t* x1_bf  = (bf16_t*)alloc((size_t)EN_ * D_ * 2);
  bf16_t* h1_b   = (bf16_t*)alloc((size_t)E_ * N_ * F_ * 2);
  float*  eo_f   = (float*) alloc((size_t)EN_ * D_ * 4);
  bf16_t* eo_bf  = (bf16_t*)alloc((size_t)EN_ * D_ * 2);
  float*  Smem   = (float*) alloc((size_t)E_ * L_ * N_ * 4);
  float*  pooled = (float*) alloc((size_t)E_ * D_ * 4);
  // aliases (non-overlapping lifetimes):
  float*  tmp_f   = ln_f;     // ln_f dead after cross_apply; tmp used by o-proj/ffn2
  bf16_t* attno_b = ln_bf;    // ln_bf dead after cross attention
  bf16_t* cat_b   = h1_b;     // h1 dead after ffn2 (cat needs 25.2MB <= 33.5MB)
  float*  out_f   = ei_f;     // ei_f dead after LN1 residual

  // 1) ln_tok = LN(tokens), f32 + bf16
  k_layernorm<<<EN_, 256, 0, stream>>>(tokens, nullptr, nullptr, nullptr, 1, 0, ln_f, ln_bf);

  // 2) weight / memory conversions to bf16
  k_cvt_bf16<<<1024, 256, 0, stream>>>(wqkv,     wqkv_b, (long)E_ * 3 * D_ * D_);
  k_cvt_bf16<<<1024, 256, 0, stream>>>(wo,       wo_b,   (long)E_ * D_ * D_);
  k_cvt_bf16<<<1024, 256, 0, stream>>>(w1,       w1_b,   (long)E_ * F_ * D_);
  k_cvt_bf16<<<1024, 256, 0, stream>>>(w2,       w2_b,   (long)E_ * D_ * F_);
  k_cvt_bf16<<<1024, 256, 0, stream>>>(agg_w,    aggw_b, (long)D_ * L_ * D_);
  k_cvt_bf16<<<1024, 256, 0, stream>>>(memories, mem_b,  (long)L_ * M_ * D_);

  // 3) cross attention (clip -> no-max softmax), two passes
  k_cross_denom<<<EN_ / 16 / 8, 256, 0, stream>>>(ln_bf, Scr);
  k_cross_apply<<<EN_ / 16, 256, 0, stream>>>(ln_bf, Scr, ln_f, ei_f, ei_bf);

  // 4) encoder: QKV GEMMs (batched over experts)
  for (int j = 0; j < 3; ++j) {
    bf16_t* dst = (j == 0) ? q_b : (j == 1) ? k_b : v_b;
    k_gemm<<<dim3(N_ / 32, D_ / 64, E_), 256, 0, stream>>>(
        ei_bf, D_, (long)N_ * D_,
        wqkv_b + (long)j * D_ * D_, D_, (long)3 * D_ * D_,
        bqkv + (long)j * D_, (long)3 * D_,
        nullptr, 0, nullptr, dst, D_, (long)N_ * D_, D_, 0);
  }

  // 5) multi-head self-attention
  k_self_denom<<<dim3(N_ / 16 / 8, E_ * H_), 256, 0, stream>>>(q_b, k_b, Sat);
  k_self_apply<<<dim3(N_ / 16, E_ * H_), 128, 0, stream>>>(q_b, k_b, v_b, Sat, attno_b);

  // 6) O-projection, residual + LN1
  k_gemm<<<dim3(N_ / 32, D_ / 64, E_), 256, 0, stream>>>(
      attno_b, D_, (long)N_ * D_, wo_b, D_, (long)D_ * D_,
      bo, D_, nullptr, 0, tmp_f, nullptr, D_, (long)N_ * D_, D_, 0);
  k_layernorm<<<EN_, 256, 0, stream>>>(tmp_f, ei_f, ln_g, ln_b, N_, (long)2 * D_, x1_f, x1_bf);

  // 7) FFN + residual + LN2
  k_gemm<<<dim3(N_ / 32, F_ / 64, E_), 256, 0, stream>>>(
      x1_bf, D_, (long)N_ * D_, w1_b, D_, (long)F_ * D_,
      b1, F_, nullptr, 0, nullptr, h1_b, F_, (long)N_ * F_, D_, 1);
  k_gemm<<<dim3(N_ / 32, D_ / 64, E_), 256, 0, stream>>>(
      h1_b, F_, (long)N_ * F_, w2_b, F_, (long)D_ * F_,
      b2, D_, nullptr, 0, tmp_f, nullptr, D_, (long)N_ * D_, F_, 0);
  k_layernorm<<<EN_, 256, 0, stream>>>(tmp_f, x1_f, ln_g + D_, ln_b + D_, N_, (long)2 * D_,
                                       eo_f, eo_bf);

  // 8) hierarchical memory retrieval (two passes) + agg projection
  k_mem_denom<<<dim3(N_ / 16 / 8, E_ * L_), 256, 0, stream>>>(eo_bf, mem_b, Smem);
  k_mem_apply<<<dim3(N_ / 16, E_ * L_), 256, 0, stream>>>(eo_bf, mem_b, Smem, cat_b);
  k_gemm<<<dim3(EN_ / 32, D_ / 64, 1), 256, 0, stream>>>(
      cat_b, L_ * D_, 0, aggw_b, L_ * D_, 0,
      agg_b, 0, eo_f, 0, out_f, nullptr, D_, 0, L_ * D_, 0);

  // 9) memory update -> new_mem (directly into d_out)
  k_mem_update<<<dim3(M_ / 16, L_), 256, 0, stream>>>(mem_b, memories, eo_bf, Smem, out_newmem);

  // 10) gating + fusion
  k_pooled<<<(E_ * D_ + 255) / 256, 256, 0, stream>>>(out_f, pooled);
  k_gate<<<1, 256, 0, stream>>>(pooled, gate_w, out_gate);
  k_fused<<<1024, 256, 0, stream>>>(out_f, out_gate, out_fused);
}
